// MGU_12343736008987
// MI455X (gfx1250) — matrix operand order repaired and checked by reading, not compiled, not run on hardware
//
#include <hip/hip_runtime.h>

typedef __bf16 bf16;
typedef __attribute__((ext_vector_type(16))) __bf16 v16bf;
typedef __attribute__((ext_vector_type(8)))  __bf16 v8bf;
typedef __attribute__((ext_vector_type(8)))  float  v8f;

// Problem constants (match reference): T=512, B=64, I=H=1024, L=2
#define TT 512
#define BB 64
#define HH 1024
#define NSTATE (BB * HH)            // 65536 elements per [B,H] state
#define CHUNK_ELEMS 512             // one packed K-chunk: 2 halves * 32 lanes * 8 elems
#define TILE_ELEMS  16384           // one packed 16-row tile: 32 chunks * 512

struct ABfrag { union { v16bf v; v8bf h[2]; }; };

__device__ __forceinline__ bf16 f2bf(float f) {
  union { float f; unsigned u; } c; c.f = f;
  unsigned r = c.u + 0x7FFFu + ((c.u >> 16) & 1u);   // round-to-nearest-even
  unsigned short hv = (unsigned short)(r >> 16);
  bf16 o; __builtin_memcpy(&o, &hv, sizeof(o));
  return o;
}

// ---------------------------------------------------------------------------
// Packing kernels: fp32 -> bf16 in exact WMMA fragment order so every A/B
// fragment load in the GEMM is one coalesced 512B global_load_b128 per wave.
//
// A-matrix (activations, 16x32 16-bit layout): lane holds row m=lane&15;
//   K within chunk: kk = half*16 + (lane>>4)*8 + elem
// B-matrix (weights as W^T columns, 32x16 layout): lane holds col n=lane&15;
//   K within chunk: kk = half*16 + (lane>>4)*8 + elem
// Packed flat index: (((tile*32 + kchunk)*2 + half)*32 + lane)*8 + elem
// ---------------------------------------------------------------------------

// x: [T=512, B=64, K=1024] -> packed per t: [4 mtiles][32 kc][2][32][8]
__global__ void pack_x(const float* __restrict__ in, bf16* __restrict__ out) {
  const int gid  = blockIdx.x * blockDim.x + threadIdx.x;   // 0 .. 512*8192-1
  const int lane = gid & 31;
  const int half = (gid >> 5) & 1;
  const int kc   = (gid >> 6) & 31;
  const int mt   = (gid >> 11) & 3;
  const int t    = gid >> 13;
  const int row  = mt * 16 + (lane & 15);
  const int col  = kc * 32 + half * 16 + ((lane >> 4) << 3);
  const v8f v = *(const v8f*)(in + ((size_t)t * BB + row) * HH + col);
  v8bf o;
#pragma unroll
  for (int e = 0; e < 8; ++e) o[e] = f2bf(v[e]);
  *(v8bf*)(out + (size_t)gid * 8) = o;
}

// W: [2 layers * 2048 rows, 1024] -> packed [256 rtiles][32 kc][2][32][8]
__global__ void pack_w(const float* __restrict__ in, bf16* __restrict__ out) {
  const int gid  = blockIdx.x * blockDim.x + threadIdx.x;   // 0 .. 256*2048-1
  const int lane = gid & 31;
  const int half = (gid >> 5) & 1;
  const int kc   = (gid >> 6) & 31;
  const int rt   = gid >> 11;                               // 0..255
  const int row  = rt * 16 + (lane & 15);
  const int col  = kc * 32 + half * 16 + ((lane >> 4) << 3);
  const v8f v = *(const v8f*)(in + (size_t)row * HH + col);
  v8bf o;
#pragma unroll
  for (int e = 0; e < 8; ++e) o[e] = f2bf(v[e]);
  *(v8bf*)(out + (size_t)gid * 8) = o;
}

// h0: [2, 64, 1024] -> per-layer packed bf16 (buf 0, A-layout) + fp32 natural
__global__ void init_h(const float* __restrict__ h0,
                       bf16* __restrict__ h0p, float* __restrict__ h0f,
                       bf16* __restrict__ h1p, float* __restrict__ h1f) {
  const int gid  = blockIdx.x * blockDim.x + threadIdx.x;   // 0 .. 2*8192-1
  const int lane = gid & 31;
  const int half = (gid >> 5) & 1;
  const int kc   = (gid >> 6) & 31;
  const int mt   = (gid >> 11) & 3;
  const int l    = gid >> 13;
  const int row  = mt * 16 + (lane & 15);
  const int col  = kc * 32 + half * 16 + ((lane >> 4) << 3);
  const v8f v = *(const v8f*)(h0 + ((size_t)l * BB + row) * HH + col);
  v8bf o;
#pragma unroll
  for (int e = 0; e < 8; ++e) o[e] = f2bf(v[e]);
  bf16*  hp = l ? h1p : h0p;
  float* hf = l ? h1f : h0f;
  *(v8bf*)(hp + (size_t)(gid & 8191) * 8) = o;
  *(v8f*)(hf + (size_t)row * HH + col) = v;
}

// ---------------------------------------------------------------------------
// One MGU layer step for one wave's 16x16 h-tile, fully fused.
//  - dual independent K-chains per gate double WMMA ILP
//  - triple-buffered fragments with prefetch distance 2: group g+2's loads
//    issue before group g's WMMAs, so waits become partial (loadcnt<=24)
//    instead of full drains, hiding L2 latency inside a single wave.
// ---------------------------------------------------------------------------
__device__ __forceinline__ void mgu_layer(
    const bf16*  __restrict__ xinP,   // packed [4][32][2][32][8] layer input
    const bf16*  __restrict__ hinP,   // packed h_prev
    const float* __restrict__ hinF,   // natural fp32 h_prev
    const bf16*  __restrict__ WihP,   // packed [128][32][2][32][8] (this layer)
    const bf16*  __restrict__ WhhP,
    const float* __restrict__ bih, const float* __restrict__ bhh, // [2048]
    float* __restrict__ houtF, bf16* __restrict__ houtP,
    float* __restrict__ out_opt,
    int wid, int lane)
{
  const int mt = wid >> 6;            // batch-row tile 0..3
  const int nt = wid & 63;            // hidden-col tile 0..63
  const size_t lo = (size_t)lane * 8;

  const bf16* apx  = xinP + (size_t)mt * TILE_ELEMS + lo;
  const bf16* aph  = hinP + (size_t)mt * TILE_ELEMS + lo;
  const bf16* bxf  = WihP + (size_t)nt * TILE_ELEMS + lo;          // f-gate rows
  const bf16* bxh  = WihP + (size_t)(64 + nt) * TILE_ELEMS + lo;   // h~-gate rows
  const bf16* bhf  = WhhP + (size_t)nt * TILE_ELEMS + lo;
  const bf16* bhh_ = WhhP + (size_t)(64 + nt) * TILE_ELEMS + lo;

  v8f cf0 = {0.f,0.f,0.f,0.f,0.f,0.f,0.f,0.f};
  v8f cf1 = {0.f,0.f,0.f,0.f,0.f,0.f,0.f,0.f};
  v8f ch0 = {0.f,0.f,0.f,0.f,0.f,0.f,0.f,0.f};
  v8f ch1 = {0.f,0.f,0.f,0.f,0.f,0.f,0.f,0.f};

  // 32 groups of 2 K-chunks: groups 0..15 = phase 0 (Wih, layer input),
  // groups 16..31 = phase 1 (Whh, h_prev). Triple-buffered fragment slots.
  ABfrag av0[3], av1[3], fv0[3], fv1[3], hv0[3], hv1[3];

  auto loadg = [&](int g, int slot) {
    const int ph = g >> 4;
    const size_t co = (size_t)((g & 15) * 2) * CHUNK_ELEMS;
    const bf16* A0 = (ph ? aph  : apx) + co;
    const bf16* F0 = (ph ? bhf  : bxf) + co;
    const bf16* H0 = (ph ? bhh_ : bxh) + co;
    av0[slot].h[0] = *(const v8bf*)(A0);        av0[slot].h[1] = *(const v8bf*)(A0 + 256);
    av1[slot].h[0] = *(const v8bf*)(A0 + 512);  av1[slot].h[1] = *(const v8bf*)(A0 + 768);
    fv0[slot].h[0] = *(const v8bf*)(F0);        fv0[slot].h[1] = *(const v8bf*)(F0 + 256);
    fv1[slot].h[0] = *(const v8bf*)(F0 + 512);  fv1[slot].h[1] = *(const v8bf*)(F0 + 768);
    hv0[slot].h[0] = *(const v8bf*)(H0);        hv0[slot].h[1] = *(const v8bf*)(H0 + 256);
    hv1[slot].h[0] = *(const v8bf*)(H0 + 512);  hv1[slot].h[1] = *(const v8bf*)(H0 + 768);
  };

  loadg(0, 0);
  loadg(1, 1);
#pragma unroll
  for (int g = 0; g < 32; ++g) {
    const int cur = g % 3;
    if (g < 30) loadg(g + 2, (g + 2) % 3);     // prefetch distance 2
    cf0 = __builtin_amdgcn_wmma_f32_16x16x32_bf16(false, av0[cur].v, false, fv0[cur].v,
                                                  (short)0, cf0, false, false);
    ch0 = __builtin_amdgcn_wmma_f32_16x16x32_bf16(false, av0[cur].v, false, hv0[cur].v,
                                                  (short)0, ch0, false, false);
    cf1 = __builtin_amdgcn_wmma_f32_16x16x32_bf16(false, av1[cur].v, false, fv1[cur].v,
                                                  (short)0, cf1, false, false);
    ch1 = __builtin_amdgcn_wmma_f32_16x16x32_bf16(false, av1[cur].v, false, hv1[cur].v,
                                                  (short)0, ch1, false, false);
  }

  const v8f cf = cf0 + cf1;
  const v8f ch = ch0 + ch1;

  // Fused gate epilogue. C/D layout: elem r -> row mt*16 + r + 8*(lane>>4),
  // col nt*16 + (lane&15).
  const int l15 = lane & 15;
  const int hi  = lane >> 4;
  const int col = nt * 16 + l15;
  const int rbase = mt * 16 + hi * 8;
  const float bias_f = bih[col] + bhh[col];
  const float bias_h = bih[HH + col] + bhh[HH + col];

  // Packed-A scatter coordinates for h_next (constant per lane except row).
  const int kcC  = col >> 5;
  const int kk   = col & 31;
  const int halfC = kk >> 4;
  const int hi2  = (kk >> 3) & 1;
  const int eC   = kk & 7;
  const size_t pbase = (((size_t)(mt * 32 + kcC) * 2 + halfC) * 256)
                     + (size_t)hi2 * 128 + (size_t)hi * 64 + eC;

#pragma unroll
  for (int r = 0; r < 8; ++r) {
    const int idx = (rbase + r) * HH + col;
    const float fg = 1.0f / (1.0f + __expf(-(cf[r] + bias_f)));
    const float ht = tanhf(ch[r] + bias_h);
    const float hp = hinF[idx];
    const float hn = hp + fg * (ht - hp);        // (1-f)*h + f*tanh
    houtF[idx] = hn;
    houtP[pbase + (size_t)r * 8] = f2bf(hn);
    if (out_opt) out_opt[idx] = hn;
  }
}

// Layer-pipelined step: blocks 0..63 run layer 0 at time t, blocks 64..127
// run layer 1 at time t-1 (independent work) -> 513 sequential kernels
// instead of 1024, 512 waves per kernel.
__launch_bounds__(128)
__global__ void mgu_step2(
    int do0,
    const bf16* __restrict__ x0p, const bf16* __restrict__ h0p_in,
    const float* __restrict__ h0f_in,
    float* __restrict__ h0f_out, bf16* __restrict__ h0p_out,
    int do1,
    const bf16* __restrict__ x1p, const bf16* __restrict__ h1p_in,
    const float* __restrict__ h1f_in,
    float* __restrict__ h1f_out, bf16* __restrict__ h1p_out,
    float* __restrict__ out1,
    const bf16* __restrict__ Wih0, const bf16* __restrict__ Whh0,
    const float* __restrict__ bih0, const float* __restrict__ bhh0,
    const bf16* __restrict__ Wih1, const bf16* __restrict__ Whh1,
    const float* __restrict__ bih1, const float* __restrict__ bhh1)
{
  const int lane  = threadIdx.x & 31;
  const int wid   = ((blockIdx.x & 63) << 2) | (threadIdx.x >> 5);  // 0..255
  const int layer = blockIdx.x >> 6;
  if (layer == 0) {
    if (!do0) return;
    mgu_layer(x0p, h0p_in, h0f_in, Wih0, Whh0, bih0, bhh0,
              h0f_out, h0p_out, nullptr, wid, lane);
  } else {
    if (!do1) return;
    mgu_layer(x1p, h1p_in, h1f_in, Wih1, Whh1, bih1, bhh1,
              h1f_out, h1p_out, out1, wid, lane);
  }
}

extern "C" void kernel_launch(void* const* d_in, const int* in_sizes, int n_in,
                              void* d_out, int out_size, void* d_ws, size_t ws_size,
                              hipStream_t stream)
{
  const float* x   = (const float*)d_in[0];  // [512,64,1024]
  const float* h0  = (const float*)d_in[1];  // [2,64,1024]
  const float* Wih = (const float*)d_in[2];  // [2,2048,1024]
  const float* Whh = (const float*)d_in[3];  // [2,2048,1024]
  const float* bih = (const float*)d_in[4];  // [2,2048]
  const float* bhh = (const float*)d_in[5];  // [2,2048]
  float* out = (float*)d_out;

  // Workspace layout (~86 MB)
  char* ws = (char*)d_ws;
  size_t off = 0;
  bf16* xp    = (bf16*)(ws + off); off += (size_t)TT * NSTATE * 2;       // 64 MB
  bf16* WihP  = (bf16*)(ws + off); off += (size_t)2 * 2 * HH * HH * 2;   // 8 MB
  bf16* WhhP  = (bf16*)(ws + off); off += (size_t)2 * 2 * HH * HH * 2;   // 8 MB
  float* h0f  = (float*)(ws + off); off += (size_t)2 * NSTATE * 4;       // dbl buf
  float* h1f  = (float*)(ws + off); off += (size_t)2 * NSTATE * 4;
  bf16*  h0p  = (bf16*) (ws + off); off += (size_t)2 * NSTATE * 2;
  bf16*  h1p  = (bf16*) (ws + off); off += (size_t)2 * NSTATE * 2;

  // One-time (per launch) packing into WMMA fragment order
  pack_x<<<(TT * NSTATE / 8) / 256, 256, 0, stream>>>(x, xp);            // 16384 blocks
  pack_w<<<((size_t)2 * 2 * HH * HH / 8) / 256, 256, 0, stream>>>(Wih, WihP);
  pack_w<<<((size_t)2 * 2 * HH * HH / 8) / 256, 256, 0, stream>>>(Whh, WhhP);
  init_h<<<(2 * NSTATE / 8) / 256, 256, 0, stream>>>(h0, h0p, h0f, h1p, h1f);

  const size_t WLAYER = (size_t)2 * HH * HH;   // 2048*1024 packed elems per layer

  // Pipelined schedule: kernel k does layer0(t=k) and layer1(t=k-1).
  // h0 state: kernel k reads buf[k&1], writes buf[(k+1)&1] (init in buf 0).
  // h1 state: j = k-1 -> reads buf[(k-1)&1], writes buf[k&1] (init in buf 0).
  for (int k = 0; k <= TT; ++k) {
    const int do0 = (k < TT);
    const int do1 = (k >= 1);
    const int t0 = do0 ? k : 0;
    const int t1 = do1 ? (k - 1) : 0;
    const int b0r = k & 1, b0w = (k + 1) & 1;
    const int b1r = (k + 1) & 1, b1w = k & 1;   // (k-1)&1 == (k+1)&1
    mgu_step2<<<128, 128, 0, stream>>>(
        do0,
        xp + (size_t)t0 * NSTATE,
        h0p + (size_t)b0r * NSTATE, h0f + (size_t)b0r * NSTATE,
        h0f + (size_t)b0w * NSTATE, h0p + (size_t)b0w * NSTATE,
        do1,
        h0p + (size_t)b0r * NSTATE,              // layer-0 output h0(t=k-1)
        h1p + (size_t)b1r * NSTATE, h1f + (size_t)b1r * NSTATE,
        h1f + (size_t)b1w * NSTATE, h1p + (size_t)b1w * NSTATE,
        out + (size_t)t1 * NSTATE,
        WihP, WhhP, bih, bhh,
        WihP + WLAYER, WhhP + WLAYER, bih + 2 * HH, bhh + 2 * HH);
  }

  // h_final: layer0 written by kernel 511 -> buf[0]; layer1 by kernel 512 -> buf[0]
  hipMemcpyAsync(out + (size_t)TT * NSTATE, h0f,
                 (size_t)NSTATE * sizeof(float), hipMemcpyDeviceToDevice, stream);
  hipMemcpyAsync(out + (size_t)TT * NSTATE + NSTATE, h1f,
                 (size_t)NSTATE * sizeof(float), hipMemcpyDeviceToDevice, stream);
}
